// Caption_Model_35588099014669
// MI455X (gfx1250) — matrix operand order repaired and checked by reading, not compiled, not run on hardware
//
#include <hip/hip_runtime.h>
#include <cstdint>
#include <cstddef>

#define BATCH  64
#define DL     1000
#define DIMG   2048
#define DATT   512
#define DEMB   1000
#define NF     36
#define DICTN  9956
#define TSTEPS 20
#define NSTEP  (TSTEPS - 1)

// Padded K dims (multiples of 32 for WMMA k-steps and 32B-aligned loads)
#define K1     5056   // [h2(1000) | v_mean(2048) | word_emb(1000) | h1(1000) | pad(8)]
#define K2     4064   // [h1(1000) | v_hat(2048)  | h2(1000) | pad(16)]
#define KH     1024   // padded 1000 -> 1024 (h1pad / h2pad rows)
#define MG     4000   // 4*DL gate rows (valid)
#define MGP    4096   // gate rows padded to 8 waves * 16
#define MOUTP  9984   // DICTN padded to 128 (78 blocks * 8 waves * 16)

// GEMM tiling
#define KC     256    // K-chunk staged in LDS (multiple of 32)
#define NT     2      // 16-col batch tiles per wave; block covers 32 batch cols
#define RROWS  (16 * NT)         // 32 batch rows staged per block
#define TPR    (256 / RROWS)     // 8 copy threads per row
#define LPAD   8                 // LDS row pad (bf16 elems) -> conflict-free ds_load

typedef __attribute__((ext_vector_type(16))) __bf16 v16bf;
typedef __attribute__((ext_vector_type(8)))  __bf16 v8bf;
typedef __attribute__((ext_vector_type(8)))  float  v8f;
typedef __attribute__((ext_vector_type(4)))  float  v4f;

__device__ __forceinline__ __bf16 f2bf(float f) {
  union { float f; unsigned u; } s; s.f = f;
  unsigned r = s.u + 0x7FFFu + ((s.u >> 16) & 1u);   // round-to-nearest-even
  unsigned short h = (unsigned short)(r >> 16);
  __bf16 b; __builtin_memcpy(&b, &h, sizeof(h));
  return b;
}

// ---------------------------------------------------------------------------
// Weight prep: dst[m][k] (bf16, K padded) = concat(A[m][0:ka], B[m][0:kb], 0)
// Rows >= mvalid are zero-filled (M padded so no wave needs an early-out).
// ---------------------------------------------------------------------------
__global__ void k_buildW(const float* __restrict__ A, int ka,
                         const float* __restrict__ Bm, int kb,
                         int kpad, int mvalid, long long total,
                         __bf16* __restrict__ dst) {
  long long i = (long long)blockIdx.x * blockDim.x + threadIdx.x;
  if (i >= total) return;
  int k = (int)(i % kpad);
  int m = (int)(i / kpad);
  float v = 0.f;
  if (m < mvalid) {
    if (k < ka)            v = A[(long long)m * ka + k];
    else if (k < ka + kb)  v = Bm[(long long)m * kb + (k - ka)];
  }
  dst[i] = f2bf(v);
}

// ---------------------------------------------------------------------------
// One-time init: v_mean, broadcast h1_0/h2_0/c1_0/c2_0, zero K-pads
// ---------------------------------------------------------------------------
__global__ void k_init(const float* __restrict__ feats,
                       const float* __restrict__ h1_0, const float* __restrict__ c1_0,
                       const float* __restrict__ h2_0, const float* __restrict__ c2_0,
                       __bf16* __restrict__ xcat1, __bf16* __restrict__ xcat2,
                       __bf16* __restrict__ h1pad, __bf16* __restrict__ h2pad,
                       float* __restrict__ c1, float* __restrict__ c2) {
  const int b = blockIdx.x, tid = threadIdx.x;
  for (int f = tid; f < DIMG; f += blockDim.x) {
    float s = 0.f;
    for (int n = 0; n < NF; ++n) s += feats[((size_t)b * NF + n) * DIMG + f];
    xcat1[(size_t)b * K1 + DL + f] = f2bf(s * (1.0f / NF));   // v_mean (time-invariant)
  }
  for (int j = tid; j < DL; j += blockDim.x) {
    float h1v = h1_0[j], h2v = h2_0[j];
    xcat1[(size_t)b * K1 + j]                    = f2bf(h2v);  // x1: h2 slot
    xcat1[(size_t)b * K1 + DL + DIMG + DEMB + j] = f2bf(h1v);  // x1: h1 (hh) slot
    xcat2[(size_t)b * K2 + j]                    = f2bf(h1v);  // x2: h1 slot
    xcat2[(size_t)b * K2 + DL + DIMG + j]        = f2bf(h2v);  // x2: h2 (hh) slot
    c1[(size_t)b * DL + j] = c1_0[j];
    c2[(size_t)b * DL + j] = c2_0[j];
  }
  if (tid < K1 - 5048) xcat1[(size_t)b * K1 + 5048 + tid] = f2bf(0.f);
  if (tid < K2 - 4048) xcat2[(size_t)b * K2 + 4048 + tid] = f2bf(0.f);
  if (tid < KH - DL) {
    h1pad[(size_t)b * KH + DL + tid] = f2bf(0.f);
    h2pad[(size_t)b * KH + DL + tid] = f2bf(0.f);
  }
}

// ---------------------------------------------------------------------------
// Word-embedding gather (teacher forcing): word_emb = W_embed[:, idx]
// ---------------------------------------------------------------------------
__global__ void k_embed(const float* __restrict__ W_embed,
                        const int* __restrict__ words, int t,
                        __bf16* __restrict__ xcat1) {
  const int b = blockIdx.x;
  const int idx = (t == 0) ? 1 : words[b * TSTEPS + t];   // START_IDX = 1
  for (int j = threadIdx.x; j < DEMB; j += blockDim.x)
    xcat1[(size_t)b * K1 + DL + DIMG + j] = f2bf(W_embed[(size_t)j * DICTN + idx]);
}

// ---------------------------------------------------------------------------
// WMMA GEMM: Y[n, m] = sum_k X[n, k] * W[m, k]   (bf16 in, f32 accumulate)
//  - block = 8 waves = 8 M-tiles (128 rows of W) x 32 batch columns
//  - X chunk (32 rows x KC) double-buffered in LDS, shared by all 8 waves
//  - weight (A) fragments software-pipelined 2 k-steps deep over global/L2
// ---------------------------------------------------------------------------
__device__ __forceinline__ void store_tile(v8f a, float* __restrict__ yr,
                                           int mrow, int Mvalid,
                                           const float* __restrict__ bias) {
  if (mrow + 8 <= Mvalid) {
    if (bias) {
#pragma unroll
      for (int v = 0; v < 8; ++v) a[v] += bias[mrow + v];
    }
    v4f lo = {a[0], a[1], a[2], a[3]};
    v4f hi = {a[4], a[5], a[6], a[7]};
    *(v4f*)yr       = lo;
    *(v4f*)(yr + 4) = hi;
  } else {
#pragma unroll
    for (int v = 0; v < 8; ++v) {
      int m = mrow + v;
      if (m < Mvalid) yr[v] = a[v] + (bias ? bias[m] : 0.f);
    }
  }
}

__global__ __launch_bounds__(256)
void k_wmma_gemm(const __bf16* __restrict__ W,   // [Mpad, K] row-major bf16 (Mpad % 128 == 0)
                 const __bf16* __restrict__ X,   // [N, K] row-major bf16
                 float* __restrict__ Y,          // [N, ldy] row-major f32
                 int K, int ldy, int Mvalid,
                 const float* __restrict__ bias) {
  __shared__ __align__(16) __bf16 xs[2][RROWS][KC + LPAD];

  const int tid  = threadIdx.x;
  const int lane = tid & 31;
  const int wave = tid >> 5;
  const int m0 = (blockIdx.x * 8 + wave) * 16;
  const int n0 = blockIdx.y * RROWS;
  const int hs = lane >> 4;                 // half-select (0/1)
  const int la = lane & 15;

  // A layout: lane holds row m0+la; K = {kb..kb+7} U {kb+16..kb+23}, kb = hs*8
  const __bf16* wrow = W + (size_t)(m0 + la) * K + hs * 8;

  // cooperative X staging: TPR threads per batch row
  const int r = tid / TPR, q = tid % TPR;
  const __bf16* xrow = X + (size_t)(n0 + r) * K;
  {
    const int kc0 = (K < KC) ? K : KC;
    for (int kk = q * 8; kk < kc0; kk += TPR * 8)
      *(v8bf*)&xs[0][r][kk] = *(const v8bf*)(xrow + kk);
  }

  v8f acc[NT];
#pragma unroll
  for (int j = 0; j < NT; ++j) acc[j] = (v8f){};

  // A-pipeline preload, depth 2 (k = 0 and k = 32)
  v8bf a0 = *(const v8bf*)(wrow + 0);
  v8bf a1 = *(const v8bf*)(wrow + 16);
  v8bf n0a = *(const v8bf*)(wrow + 32);
  v8bf n1a = *(const v8bf*)(wrow + 48);

  __syncthreads();

  const int nchunks = (K + KC - 1) / KC;
  int buf = 0;
  for (int c = 0; c < nchunks; ++c) {
    const int k0 = c * KC;
    const int kc = (K - k0 < KC) ? (K - k0) : KC;
    // stage chunk c+1 into the other buffer (overlaps this chunk's WMMAs)
    if (c + 1 < nchunks) {
      const int k0n = k0 + KC;
      const int kcn = (K - k0n < KC) ? (K - k0n) : KC;
      __builtin_prefetch(wrow + k0n + KC, 0, 1);   // weight stream hint
      for (int kk = q * 8; kk < kcn; kk += TPR * 8)
        *(v8bf*)&xs[buf ^ 1][r][kk] = *(const v8bf*)(xrow + k0n + kk);
    }
    // compute on current chunk; A pipelined two k-steps ahead
#pragma unroll 2
    for (int kk = 0; kk < kc; kk += 32) {
      const int kg = k0 + kk;
      v8bf f0 = *(const v8bf*)(wrow + kg + 64);   // tail over-read stays in d_ws
      v8bf f1 = *(const v8bf*)(wrow + kg + 80);
      union { v16bf v; v8bf h[2]; } a;
      a.h[0] = a0; a.h[1] = a1;
      // issue all NT B-tile LDS loads before the WMMAs
      union { v16bf v; v8bf h[2]; } bb[NT];
#pragma unroll
      for (int j = 0; j < NT; ++j) {
        bb[j].h[0] = *(const v8bf*)&xs[buf][16 * j + la][kk + hs * 8];
        bb[j].h[1] = *(const v8bf*)&xs[buf][16 * j + la][kk + 16 + hs * 8];
      }
#pragma unroll
      for (int j = 0; j < NT; ++j)
        acc[j] = __builtin_amdgcn_wmma_f32_16x16x32_bf16(false, a.v, false, bb[j].v,
                                                         (short)0, acc[j], false, false);
      a0 = n0a; a1 = n1a;
      n0a = f0; n1a = f1;
    }
    __syncthreads();
    buf ^= 1;
  }

  // D layout: lane holds N = la, M = m0 + hs*8 + v -> 8 contiguous floats
  const int mrow = m0 + hs * 8;
#pragma unroll
  for (int j = 0; j < NT; ++j)
    store_tile(acc[j], Y + (size_t)(n0 + 16 * j + la) * ldy + mrow, mrow, Mvalid, bias);
}

// ---------------------------------------------------------------------------
// LSTM pointwise: gates -> (h, c); scatter h as bf16 into next activations
// ---------------------------------------------------------------------------
__global__ void k_lstm_pw(const float* __restrict__ gates,
                          const float* __restrict__ b_ih, const float* __restrict__ b_hh,
                          float* __restrict__ c,
                          __bf16* __restrict__ dstA, int strideA,
                          __bf16* __restrict__ dstB, int strideB,
                          __bf16* __restrict__ hpad) {
  const int b = blockIdx.x;
  const float* g = gates + (size_t)b * MG;
  for (int j = threadIdx.x; j < DL; j += blockDim.x) {
    float gi = g[j]          + b_ih[j]          + b_hh[j];
    float gf = g[DL + j]     + b_ih[DL + j]     + b_hh[DL + j];
    float gg = g[2 * DL + j] + b_ih[2 * DL + j] + b_hh[2 * DL + j];
    float go = g[3 * DL + j] + b_ih[3 * DL + j] + b_hh[3 * DL + j];
    float si = 1.f / (1.f + expf(-gi));
    float sf = 1.f / (1.f + expf(-gf));
    float so = 1.f / (1.f + expf(-go));
    float cn = sf * c[(size_t)b * DL + j] + si * tanhf(gg);
    float hn = so * tanhf(cn);
    c[(size_t)b * DL + j] = cn;
    __bf16 hb = f2bf(hn);
    dstA[(size_t)b * strideA + j] = hb;
    dstB[(size_t)b * strideB + j] = hb;
    hpad[(size_t)b * KH + j] = hb;
  }
}

// ---------------------------------------------------------------------------
// Visual attention: logits = wa . tanh(img_emb + ah) + ba; softmax over 36
// regions; v_hat = attn-weighted sum of image features (written as bf16).
// ---------------------------------------------------------------------------
__global__ __launch_bounds__(256)
void k_attn(const float* __restrict__ img_emb,  // [B*NF, DATT]
            const float* __restrict__ ah,       // [B, DATT]
            const float* __restrict__ wa, const float* __restrict__ ba,
            const float* __restrict__ feats,    // [B, NF, DIMG]
            __bf16* __restrict__ vhat_dst) {    // xcat2 + DL
  __shared__ float ahs[DATT];
  __shared__ float red[256];
  __shared__ float logits[NF];
  __shared__ float attn[NF];
  const int b = blockIdx.x, tid = threadIdx.x;
  for (int a = tid; a < DATT; a += blockDim.x) ahs[a] = ah[(size_t)b * DATT + a];
  __syncthreads();
  for (int n = 0; n < NF; ++n) {
    const float* e = img_emb + ((size_t)b * NF + n) * DATT;
    float p = 0.f;
    for (int a = tid; a < DATT; a += blockDim.x) p += tanhf(e[a] + ahs[a]) * wa[a];
    red[tid] = p; __syncthreads();
    for (int s = 128; s > 0; s >>= 1) {
      if (tid < s) red[tid] += red[tid + s];
      __syncthreads();
    }
    if (tid == 0) logits[n] = red[0] + ba[0];
    __syncthreads();
  }
  float mx = logits[0];
  for (int n = 1; n < NF; ++n) mx = fmaxf(mx, logits[n]);
  float ssum = 0.f;
  for (int n = 0; n < NF; ++n) ssum += expf(logits[n] - mx);
  if (tid < NF) attn[tid] = expf(logits[tid] - mx) / ssum;
  __syncthreads();
  for (int f = tid; f < DIMG; f += blockDim.x) {
    float acc = 0.f;
    for (int n = 0; n < NF; ++n) acc += attn[n] * feats[((size_t)b * NF + n) * DIMG + f];
    vhat_dst[(size_t)b * K2 + f] = f2bf(acc);
  }
}

// ---------------------------------------------------------------------------
extern "C" void kernel_launch(void* const* d_in, const int* in_sizes, int n_in,
                              void* d_out, int out_size, void* d_ws, size_t ws_size,
                              hipStream_t stream) {
  (void)in_sizes; (void)n_in; (void)out_size; (void)ws_size;
  const float* feats   = (const float*)d_in[0];
  const int*   words   = (const int*)d_in[2];
  const float* W_embed = (const float*)d_in[3];
  const float* W1_ih   = (const float*)d_in[4];
  const float* W1_hh   = (const float*)d_in[5];
  const float* b1_ih   = (const float*)d_in[6];
  const float* b1_hh   = (const float*)d_in[7];
  const float* W2_ih   = (const float*)d_in[8];
  const float* W2_hh   = (const float*)d_in[9];
  const float* b2_ih   = (const float*)d_in[10];
  const float* b2_hh   = (const float*)d_in[11];
  const float* Wa_img  = (const float*)d_in[12];
  const float* Wa_h    = (const float*)d_in[13];
  const float* wa      = (const float*)d_in[14];
  const float* ba      = (const float*)d_in[15];
  const float* Wp      = (const float*)d_in[16];
  const float* bp      = (const float*)d_in[17];
  const float* h1_0    = (const float*)d_in[18];
  const float* c1_0    = (const float*)d_in[19];
  const float* h2_0    = (const float*)d_in[20];
  const float* c2_0    = (const float*)d_in[21];
  float* out = (float*)d_out;

  size_t off = 0;
  auto carve = [&](size_t bytes) -> char* {
    char* p = (char*)d_ws + off;
    off += (bytes + 255) & ~(size_t)255;
    return p;
  };
  __bf16* W1cat   = (__bf16*)carve((size_t)MGP * K1 * 2);          // [4096, 5056]
  __bf16* W2cat   = (__bf16*)carve((size_t)MGP * K2 * 2);          // [4096, 4064]
  __bf16* Wpcat   = (__bf16*)carve((size_t)MOUTP * KH * 2);        // [9984, 1024]
  __bf16* Wahc    = (__bf16*)carve((size_t)DATT * KH * 2);         // [512, 1024]
  __bf16* Waic    = (__bf16*)carve((size_t)DATT * DIMG * 2);       // [512, 2048]
  __bf16* featsbf = (__bf16*)carve((size_t)BATCH * NF * DIMG * 2); // [2304, 2048]
  float*  imgemb  = (float*)carve((size_t)BATCH * NF * DATT * 4);  // [2304, 512]
  __bf16* xcat1   = (__bf16*)carve((size_t)BATCH * K1 * 2);
  __bf16* xcat2   = (__bf16*)carve((size_t)BATCH * K2 * 2);
  __bf16* h1pad   = (__bf16*)carve((size_t)BATCH * KH * 2);
  __bf16* h2pad   = (__bf16*)carve((size_t)BATCH * KH * 2);
  float*  ah      = (float*)carve((size_t)BATCH * DATT * 4);
  float*  gates   = (float*)carve((size_t)BATCH * MG * 4);
  float*  c1      = (float*)carve((size_t)BATCH * DL * 4);
  float*  c2      = (float*)carve((size_t)BATCH * DL * 4);
  (void)carve(256);   // tail pad: A-pipeline over-read safety

  auto gridB = [](long long total) { return dim3((unsigned)((total + 255) / 256)); };

  // One-time bf16 weight prep (ih|hh fused; K padded to 32, M padded to 128)
  k_buildW<<<gridB((long long)MGP * K1), 256, 0, stream>>>(W1_ih, 4048, W1_hh, DL, K1, MG, (long long)MGP * K1, W1cat);
  k_buildW<<<gridB((long long)MGP * K2), 256, 0, stream>>>(W2_ih, 3048, W2_hh, DL, K2, MG, (long long)MGP * K2, W2cat);
  k_buildW<<<gridB((long long)MOUTP * KH), 256, 0, stream>>>(Wp, DL, nullptr, 0, KH, DICTN, (long long)MOUTP * KH, Wpcat);
  k_buildW<<<gridB((long long)DATT * KH), 256, 0, stream>>>(Wa_h, DL, nullptr, 0, KH, DATT, (long long)DATT * KH, Wahc);
  k_buildW<<<gridB((long long)DATT * DIMG), 256, 0, stream>>>(Wa_img, DIMG, nullptr, 0, DIMG, DATT, (long long)DATT * DIMG, Waic);
  k_buildW<<<gridB((long long)BATCH * NF * DIMG), 256, 0, stream>>>(feats, DIMG, nullptr, 0, DIMG, BATCH * NF, (long long)BATCH * NF * DIMG, featsbf);

  k_init<<<BATCH, 256, 0, stream>>>(feats, h1_0, c1_0, h2_0, c2_0, xcat1, xcat2, h1pad, h2pad, c1, c2);

  // Time-invariant: img_emb[bn, a] = feats[bn, :] . Wa_img[a, :]
  k_wmma_gemm<<<dim3(512 / 128, (BATCH * NF) / RROWS), 256, 0, stream>>>(
      Waic, featsbf, imgemb, DIMG, DATT, DATT, nullptr);

  for (int t = 0; t < NSTEP; ++t) {
    k_embed<<<BATCH, 256, 0, stream>>>(W_embed, words, t, xcat1);
    // gates1 = xcat1 @ W1cat^T
    k_wmma_gemm<<<dim3(MGP / 128, BATCH / RROWS), 256, 0, stream>>>(
        W1cat, xcat1, gates, K1, MG, MG, nullptr);
    k_lstm_pw<<<BATCH, 256, 0, stream>>>(gates, b1_ih, b1_hh, c1,
                                         xcat2, K2,                       // h1 -> x2 slot
                                         xcat1 + (DL + DIMG + DEMB), K1,  // h1 -> x1 hh slot
                                         h1pad);
    // ah = h1 @ Wa_h^T
    k_wmma_gemm<<<dim3(512 / 128, BATCH / RROWS), 256, 0, stream>>>(
        Wahc, h1pad, ah, KH, DATT, DATT, nullptr);
    k_attn<<<BATCH, 256, 0, stream>>>(imgemb, ah, wa, ba, feats, xcat2 + DL);
    // gates2 = xcat2 @ W2cat^T
    k_wmma_gemm<<<dim3(MGP / 128, BATCH / RROWS), 256, 0, stream>>>(
        W2cat, xcat2, gates, K2, MG, MG, nullptr);
    k_lstm_pw<<<BATCH, 256, 0, stream>>>(gates, b2_ih, b2_hh, c2,
                                         xcat1, K1,                       // h2 -> x1 slot
                                         xcat2 + (DL + DIMG), K2,         // h2 -> x2 hh slot
                                         h2pad);
    // y[:, t, :] = h2 @ Wp^T + bp
    k_wmma_gemm<<<dim3(MOUTP / 128, BATCH / RROWS), 256, 0, stream>>>(
        Wpcat, h2pad, out + (size_t)t * DICTN, KH, NSTEP * DICTN, DICTN, bp);
  }
}
